// Vanilla_RNN_19756849562114
// MI455X (gfx1250) — matrix operand order repaired and checked by reading, not compiled, not run on hardware
//
#include <hip/hip_runtime.h>
#include <hip/hip_bf16.h>

// ---------------------------------------------------------------------------
// Vanilla 2-layer tanh RNN (B=256,S=2048,F=64,I=H=128) for MI455X / gfx1250.
// enc GEMM (many WGs) -> persistent recurrence+decoder kernel (16 WGs, one per
// 16-batch-row slice), bf16 WMMA 16x16x32 with f32 accumulate.
// Async global->LDS double-buffered xe staging (ASYNCcnt) + v_tanh_f32.
// ---------------------------------------------------------------------------

typedef __attribute__((ext_vector_type(16))) __bf16 v16bf;
typedef __attribute__((ext_vector_type(8)))  __bf16 v8bf;
typedef __attribute__((ext_vector_type(8)))  float  v8f;
typedef int v4i __attribute__((vector_size(16)));

#define B_ 256
#define S_ 2048
#define F_ 64
#define I_ 128
#define H_ 128

#if defined(__AMDGCN__) && __has_builtin(__builtin_amdgcn_global_load_async_to_lds_b128) && \
    __has_builtin(__builtin_amdgcn_s_wait_asynccnt)
#define HAVE_ASYNC_LDS 1
#endif

#if defined(__AMDGCN__) && __has_builtin(__builtin_amdgcn_tanhf)
#define TANHF(x) __builtin_amdgcn_tanhf(x)
#else
#define TANHF(x) tanhf(x)
#endif

union V16U { v16bf v; v8bf h[2]; };

// A-tile (16x32 bf16, M x K) from row-major LDS [rows][ld]:
// lane l: m = l&15, half = l>>4; elems j -> k = kb + 16*half + j  (16 contiguous)
__device__ __forceinline__ v16bf loadA(const __bf16* base, int ld, int kb, int lane) {
  const int m = lane & 15, hf = lane >> 4;
  const __bf16* p = base + m * ld + kb + 16 * hf;
  V16U u;
  u.h[0] = *(const v8bf*)(p);
  u.h[1] = *(const v8bf*)(p + 8);
  return u.v;
}

// B-tile (32x16, K x N) == W^T tile: B[k][n] = W[n0+n][kb+k], W row-major f32.
// lane l: n = n0 + (l&15); elems j -> k = kb + 16*(l>>4) + j  (16 contiguous).
__device__ __forceinline__ v16bf loadBg(const float* W, int ld, int n0, int kb, int lane) {
  const int n = n0 + (lane & 15), hf = lane >> 4;
  const float* p = W + (size_t)n * ld + kb + 16 * hf;
  v16bf r;
#pragma unroll
  for (int j = 0; j < 16; ++j) r[j] = (__bf16)p[j];
  return r;
}

__device__ __forceinline__ v8f wmma_bf16(v16bf a, v16bf b, v8f c) {
  return __builtin_amdgcn_wmma_f32_16x16x32_bf16(false, a, false, b, (short)0, c,
                                                 false, false);
}

// Async 4KB tile copy: 256 threads x 16B. Global -> LDS without VGPR bounce.
__device__ __forceinline__ void tile_load_async(const __bf16* g, __bf16* l, int tid) {
#ifdef HAVE_ASYNC_LDS
  __builtin_amdgcn_global_load_async_to_lds_b128(
      (__attribute__((address_space(1))) v4i*)(g + tid * 8),
      (__attribute__((address_space(3))) v4i*)(l + tid * 8), 0, 0);
#else
  ((uint4*)l)[tid] = ((const uint4*)g)[tid];
#endif
}

template <int N>
__device__ __forceinline__ void tile_wait() {
#ifdef HAVE_ASYNC_LDS
  __builtin_amdgcn_s_wait_asynccnt(N);
#endif
}

// ---------------------------------------------------------------------------
// Encoder: xe[s][b][i] = bf16( x[b][s][:] @ enc_W^T + enc_b )   (time-major!)
// 128 rows (of B*S) per block, 8 waves: wave w owns output columns [16w,16w+16).
// ---------------------------------------------------------------------------
__global__ __launch_bounds__(256)
void enc_kernel(const float* __restrict__ x, const float* __restrict__ encW,
                const float* __restrict__ encB, __bf16* __restrict__ xe) {
  __shared__ __align__(16) __bf16 sX[128 * F_];   // 16 KB: 128 rows x 64 bf16
  __shared__ float sB[I_];

  const int tid = threadIdx.x, lane = tid & 31, wave = tid >> 5;
  const int r0 = blockIdx.x * 128;      // global row in [S*B), row = s*B + b
  const int s  = r0 / B_;
  const int b0 = r0 % B_;               // 0 or 128 (128 | 256)

  for (int i = tid; i < I_; i += 256) sB[i] = encB[i];

  // stage 128 rows x 64 f32 of x -> bf16 LDS (2 threads per row, 32 elems each)
  {
    const int row = tid >> 1, c0 = (tid & 1) * 32;
    const float* src = x + (size_t)(b0 + row) * S_ * F_ + (size_t)s * F_ + c0;
    __bf16* dst = sX + row * F_ + c0;
#pragma unroll
    for (int c = 0; c < 32; ++c) dst[c] = (__bf16)src[c];
  }

  // per-wave weight B-tiles in registers (wave only needs its 16 W rows)
  const int n0 = wave * 16;
  v16bf w0 = loadBg(encW, F_, n0, 0,  lane);
  v16bf w1 = loadBg(encW, F_, n0, 32, lane);
  __syncthreads();

  const int hf = lane >> 4, nn = lane & 15;
  __bf16* outbase = xe + (size_t)(s * B_ + b0) * I_;
#pragma unroll
  for (int rt = 0; rt < 8; ++rt) {
    v8f acc;
    const float bb = sB[n0 + nn];
#pragma unroll
    for (int r = 0; r < 8; ++r) acc[r] = bb;
    acc = wmma_bf16(loadA(sX + rt * 16 * F_, F_, 0,  lane), w0, acc);
    acc = wmma_bf16(loadA(sX + rt * 16 * F_, F_, 32, lane), w1, acc);
#pragma unroll
    for (int r = 0; r < 8; ++r) {
      const int m = rt * 16 + r + 8 * hf;                 // C/D: VGPR r -> M=r(+8)
      outbase[(size_t)m * I_ + n0 + nn] = (__bf16)acc[r];
    }
  }
}

// ---------------------------------------------------------------------------
// Persistent recurrence + fused decoder. One block per 16 batch rows (16 WGs).
// Weights live in registers (per-wave 16-row slices); h0/h1 double-buffered
// in LDS as bf16; xe tiles double-buffered via async global->LDS DMA.
// ---------------------------------------------------------------------------
__global__ __launch_bounds__(256)
void rnn_kernel(const __bf16* __restrict__ xe,
                const float* __restrict__ Wih0, const float* __restrict__ Whh0,
                const float* __restrict__ bih0, const float* __restrict__ bhh0,
                const float* __restrict__ Wih1, const float* __restrict__ Whh1,
                const float* __restrict__ bih1, const float* __restrict__ bhh1,
                const float* __restrict__ decW, const float* __restrict__ decB,
                float* __restrict__ out) {
  __shared__ __align__(16) __bf16 sH0[2][16 * H_];   // 2 x 4 KB
  __shared__ __align__(16) __bf16 sH1[2][16 * H_];   // 2 x 4 KB
  __shared__ __align__(16) __bf16 sXe[2][16 * I_];   // 2 x 4 KB, async staged
  __shared__ float sB0[H_], sB1[H_], sBd[F_];

  const int tid = threadIdx.x, lane = tid & 31, wave = tid >> 5;
  const int b0 = blockIdx.x * 16;                    // batch slice
  const int n0 = wave * 16;
  const int hf = lane >> 4, nn = lane & 15;

  // kick off tile 0 DMA immediately, before the (long) init phase
  tile_load_async(xe + (size_t)b0 * I_, sXe[0], tid);

  // combined biases + zero hidden state
  for (int i = tid; i < H_; i += 256) { sB0[i] = bih0[i] + bhh0[i];
                                        sB1[i] = bih1[i] + bhh1[i]; }
  for (int i = tid; i < F_; i += 256) sBd[i] = decB[i];
  for (int i = tid; i < 16 * H_; i += 256) {
    sH0[0][i] = (__bf16)0.f; sH0[1][i] = (__bf16)0.f;
    sH1[0][i] = (__bf16)0.f; sH1[1][i] = (__bf16)0.f;
  }

  // per-wave weight slices -> registers (used 2048 times each)
  v16bf wih0[4], whh0[4], wih1[4], whh1[4], wdec[4];
#pragma unroll
  for (int k = 0; k < 4; ++k) {
    wih0[k] = loadBg(Wih0, I_, n0, 32 * k, lane);
    whh0[k] = loadBg(Whh0, H_, n0, 32 * k, lane);
    wih1[k] = loadBg(Wih1, H_, n0, 32 * k, lane);
    whh1[k] = loadBg(Whh1, H_, n0, 32 * k, lane);
  }
  if (wave < 4) {
#pragma unroll
    for (int k = 0; k < 4; ++k) wdec[k] = loadBg(decW, H_, n0, 32 * k, lane);
  }

  int cur = 0;
  for (int t = 0; t < S_; ++t) {
    const int nxt = cur ^ 1;
    const int xb  = t & 1;

    // issue DMA for tile t+1 (overlaps the whole step-t compute), then wait
    // only for tile t (issued last iteration): asynccnt<=1. In-order per wave.
    if (t + 1 < S_) {
      tile_load_async(xe + ((size_t)(t + 1) * B_ + b0) * I_, sXe[xb ^ 1], tid);
      tile_wait<1>();
    } else {
      tile_wait<0>();
    }
    __syncthreads();   // all waves' tile-t portions + prev-step h visible

    // layer 0: h0' = tanh(xe_t @ Wih0^T + h0 @ Whh0^T + b)
    {
      v8f acc; const float bb = sB0[n0 + nn];
#pragma unroll
      for (int r = 0; r < 8; ++r) acc[r] = bb;
#pragma unroll
      for (int k = 0; k < 4; ++k)
        acc = wmma_bf16(loadA(sXe[xb], I_, 32 * k, lane), wih0[k], acc);
#pragma unroll
      for (int k = 0; k < 4; ++k)
        acc = wmma_bf16(loadA(sH0[cur], H_, 32 * k, lane), whh0[k], acc);
#pragma unroll
      for (int r = 0; r < 8; ++r)
        sH0[nxt][(r + 8 * hf) * H_ + n0 + nn] = (__bf16)TANHF(acc[r]);
    }
    __syncthreads();

    // layer 1: h1' = tanh(h0' @ Wih1^T + h1 @ Whh1^T + b)
    {
      v8f acc; const float bb = sB1[n0 + nn];
#pragma unroll
      for (int r = 0; r < 8; ++r) acc[r] = bb;
#pragma unroll
      for (int k = 0; k < 4; ++k)
        acc = wmma_bf16(loadA(sH0[nxt], H_, 32 * k, lane), wih1[k], acc);
#pragma unroll
      for (int k = 0; k < 4; ++k)
        acc = wmma_bf16(loadA(sH1[cur], H_, 32 * k, lane), whh1[k], acc);
#pragma unroll
      for (int r = 0; r < 8; ++r)
        sH1[nxt][(r + 8 * hf) * H_ + n0 + nn] = (__bf16)TANHF(acc[r]);
    }
    __syncthreads();

    // fused decoder: out[b0..b0+16, t, :] = h1' @ dec_W^T + dec_b  (waves 0..3)
    if (wave < 4) {
      v8f acc; const float bb = sBd[n0 + nn];
#pragma unroll
      for (int r = 0; r < 8; ++r) acc[r] = bb;
#pragma unroll
      for (int k = 0; k < 4; ++k)
        acc = wmma_bf16(loadA(sH1[nxt], H_, 32 * k, lane), wdec[k], acc);
#pragma unroll
      for (int r = 0; r < 8; ++r) {
        const int m = r + 8 * hf;
        out[((size_t)(b0 + m) * S_ + t) * F_ + n0 + nn] = acc[r];
      }
    }
    __syncthreads();
    cur = nxt;
  }
}

// ---------------------------------------------------------------------------
extern "C" void kernel_launch(void* const* d_in, const int* in_sizes, int n_in,
                              void* d_out, int out_size, void* d_ws, size_t ws_size,
                              hipStream_t stream) {
  (void)in_sizes; (void)n_in; (void)out_size; (void)ws_size;
  const float* x    = (const float*)d_in[0];
  const float* encW = (const float*)d_in[1];
  const float* encB = (const float*)d_in[2];
  const float* Wih0 = (const float*)d_in[3];
  const float* Whh0 = (const float*)d_in[4];
  const float* bih0 = (const float*)d_in[5];
  const float* bhh0 = (const float*)d_in[6];
  const float* Wih1 = (const float*)d_in[7];
  const float* Whh1 = (const float*)d_in[8];
  const float* bih1 = (const float*)d_in[9];
  const float* bhh1 = (const float*)d_in[10];
  const float* decW = (const float*)d_in[11];
  const float* decB = (const float*)d_in[12];
  float* out = (float*)d_out;

  __bf16* xe = (__bf16*)d_ws;   // time-major [S][B][I] bf16, 128 MB

  enc_kernel<<<(S_ * B_) / 128, 256, 0, stream>>>(x, encW, encB, xe);
  rnn_kernel<<<B_ / 16, 256, 0, stream>>>(xe, Wih0, Whh0, bih0, bhh0,
                                          Wih1, Whh1, bih1, bhh1,
                                          decW, decB, out);
}
